// SEALNetwork_10514079941184
// MI455X (gfx1250) — matrix operand order repaired and checked by reading, not compiled, not run on hardware
//
#include <hip/hip_runtime.h>

typedef __attribute__((ext_vector_type(16))) _Float16 v16h;
typedef __attribute__((ext_vector_type(8)))  float    v8f;

union Vh16 { v16h v; _Float16 e[16]; };
union Vf8  { v8f  v; float    e[8];  };

static constexpr int kG   = 256;      // graphs
static constexpr int kNPG = 128;      // nodes per graph
static constexpr int kC   = 16;       // clusters
static constexpr int kH   = 128;      // hidden
static constexpr int kL   = 3;        // layers
static constexpr int kN   = kG * kNPG;    // 32768
static constexpr int kE   = 524288;       // edges
static constexpr int kEPG = kE / kG;      // 2048 edges per graph
static constexpr int kWSZ = kL * kH * kH; // one weight tensor, flat

__device__ __forceinline__ float wave_sum(float x) {
    #pragma unroll
    for (int off = 16; off > 0; off >>= 1) x += __shfl_xor(x, off, 32);
    return x;
}

// ---- one-shot: convert Wn|Wo|Wr to f16 (row-major == WMMA-B gather layout) ----
__global__ __launch_bounds__(256)
void convert_weights(const float* __restrict__ Wn, const float* __restrict__ Wo,
                     const float* __restrict__ Wr, _Float16* __restrict__ Wh)
{
    int i = blockIdx.x * blockDim.x + threadIdx.x;
    if (i < kWSZ) {
        Wh[i]            = (_Float16)Wn[i];
        Wh[kWSZ + i]     = (_Float16)Wo[i];
        Wh[2 * kWSZ + i] = (_Float16)Wr[i];
    }
}

__global__ __launch_bounds__(256)
void seal_fused(const float* __restrict__ x,
                const long long* __restrict__ eidx,
                const unsigned char* __restrict__ maskb,
                const float* __restrict__ s,
                const float* __restrict__ bn, const float* __restrict__ bo,
                const float* __restrict__ lng, const float* __restrict__ lnb,
                const float* __restrict__ flng, const float* __restrict__ flnb,
                const float* __restrict__ linw, const float* __restrict__ bias,
                const _Float16* __restrict__ Wh,
                float* __restrict__ XI, float* __restrict__ XO,
                float* __restrict__ XR, _Float16* __restrict__ HBh,
                float* __restrict__ INV, float* __restrict__ L1P,
                float* __restrict__ out)
{
    __shared__ float AGG[kNPG * kH];          // 64 KB, multi-purpose

    const int g    = blockIdx.x;
    const int tid  = threadIdx.x;
    const int lane = tid & 31;
    const int wid  = tid >> 5;                // 8 waves
    const int lm   = lane & 15;
    const int grp  = lane >> 4;
    const int e0   = g * kEPG;
    const size_t nb = (size_t)g * kNPG;       // node base for this graph

    // ---------- phase 0: in-degree -> inverse count ----------
    if (tid < kNPG) AGG[tid] = 0.0f;
    __syncthreads();
    for (int i = tid; i < kEPG; i += 256) {
        int tl = (int)eidx[kE + e0 + i] & (kNPG - 1);
        atomicAdd(&AGG[tl], 1.0f);
    }
    __syncthreads();
    if (tid < kNPG) INV[nb + tid] = 1.0f / fmaxf(AGG[tid], 1.0f);
    __syncthreads();

    // ---------- GNN layers ----------
    for (int layer = 0; layer < kL; ++layer) {
        // ===== GEMM phase: wave `wid` owns rows [wid*16, wid*16+16) =====
        // A-fragments: 16x32 f16 tile of h (contiguous layout:
        // K = hh + grp*16 + kc*32)
        const int m = wid * 16 + lm;
        Vh16 afrag[4];
        if (layer == 0) {
            const float* arow = x + (nb + m) * kH;
            #pragma unroll
            for (int kc = 0; kc < 4; ++kc) {
                #pragma unroll
                for (int hh = 0; hh < 16; ++hh) {
                    int k = hh + grp * 16 + kc * 32;
                    afrag[kc].e[hh] = (_Float16)arow[k];
                }
            }
        } else {
            const _Float16* arow = HBh + (nb + m) * kH;
            #pragma unroll
            for (int kc = 0; kc < 4; ++kc) {
                #pragma unroll
                for (int hh = 0; hh < 16; ++hh) {
                    int k = hh + grp * 16 + kc * 32;
                    afrag[kc].e[hh] = arow[k];      // contiguous 16-half runs -> b128
                }
            }
        }

        const float* Bss[3] = { bn + layer * kH, bo + layer * kH, nullptr };
        float* Ds[3] = { XI, XO, XR };

        for (int wsel = 0; wsel < 3; ++wsel) {
            const _Float16* W = Wh + (size_t)wsel * kWSZ + (size_t)layer * kH * kH;
            #pragma unroll 2
            for (int jt = 0; jt < 8; ++jt) {
                int j = jt * 16 + lm;
                const _Float16* wrow = W + (size_t)j * kH;   // B[k][j] = W[j][k]
                Vf8 acc = {};
                #pragma unroll
                for (int kc = 0; kc < 4; ++kc) {
                    Vh16 bfrag;
                    #pragma unroll
                    for (int hh = 0; hh < 16; ++hh)          // 16 contiguous halves
                        bfrag.e[hh] = wrow[kc * 32 + grp * 16 + hh];
                    acc.v = __builtin_amdgcn_wmma_f32_16x16x32_f16(
                        false, afrag[kc].v, false, bfrag.v,
                        (short)0, acc.v, false, false);
                }
                float bb = Bss[wsel] ? Bss[wsel][j] : 0.0f;
                float* D = Ds[wsel];
                #pragma unroll
                for (int r = 0; r < 8; ++r) {
                    int row = wid * 16 + r + grp * 8;
                    D[(nb + row) * kH + j] = acc.e[r] + bb;
                }
            }
        }
        __threadfence();          // make XI/XO/XR visible across waves
        __syncthreads();

        // ===== zero aggregation tile =====
        for (int idx = tid; idx < kNPG * kH; idx += 256) AGG[idx] = 0.0f;
        __syncthreads();

        // ===== edge message pass: mean-aggregate via LDS f32 atomics =====
        for (int i = wid; i < kEPG; i += 8) {
            int e = e0 + i;
            long long src = eidx[e];
            int tl = (int)eidx[kE + e] & (kNPG - 1);
            const float* row = (maskb[e] ? XI : XO) + (size_t)src * kH;
            #pragma unroll
            for (int c4 = 0; c4 < 4; ++c4) {
                int col = lane + 32 * c4;                 // conflict-free stride-1
                atomicAdd(&AGG[tl * kH + col], row[col]);
            }
        }
        __syncthreads();

        // ===== agg*inv + x_root -> LayerNorm -> ReLU -> HBh (f16) =====
        for (int rrow = wid; rrow < kNPG; rrow += 8) {
            float inv = INV[nb + rrow];
            const float* xrr = XR + (nb + rrow) * kH;
            float v[4]; float ssum = 0.0f;
            #pragma unroll
            for (int c4 = 0; c4 < 4; ++c4) {
                int col = lane + 32 * c4;
                v[c4] = AGG[rrow * kH + col] * inv + xrr[col];
                ssum += v[c4];
            }
            float mu = wave_sum(ssum) * (1.0f / kH);
            float sq = 0.0f;
            #pragma unroll
            for (int c4 = 0; c4 < 4; ++c4) { float d = v[c4] - mu; sq += d * d; }
            float rn = rsqrtf(wave_sum(sq) * (1.0f / kH) + 1e-5f);
            _Float16* hout = HBh + (nb + rrow) * kH;
            #pragma unroll
            for (int c4 = 0; c4 < 4; ++c4) {
                int col = lane + 32 * c4;
                float o = (v[c4] - mu) * rn * lng[layer * kH + col] + lnb[layer * kH + col];
                hout[col] = (_Float16)fmaxf(o, 0.0f);
            }
        }
        __threadfence();
        __syncthreads();
    }

    // ---------- pooling: pooled = s_g^T (16x128) @ h_g (128x128), WMMA ----------
    {
        const float* sg = s + nb * kC;
        Vh16 afrag[4];                          // A[c][k] = s_g[k][c]
        #pragma unroll
        for (int kc = 0; kc < 4; ++kc) {
            #pragma unroll
            for (int hh = 0; hh < 16; ++hh) {
                int k = hh + grp * 16 + kc * 32;
                afrag[kc].e[hh] = (_Float16)sg[(size_t)k * kC + lm];
            }
        }
        int j = wid * 16 + lm;                  // wave `wid` owns N-tile `wid`
        Vf8 acc = {};
        #pragma unroll
        for (int kc = 0; kc < 4; ++kc) {
            Vh16 bfrag;                         // B[k][j] = h_g[k][j]
            #pragma unroll
            for (int hh = 0; hh < 16; ++hh) {
                int k = kc * 32 + grp * 16 + hh;
                bfrag.e[hh] = HBh[(nb + k) * kH + j];
            }
            acc.v = __builtin_amdgcn_wmma_f32_16x16x32_f16(
                false, afrag[kc].v, false, bfrag.v, (short)0, acc.v, false, false);
        }
        #pragma unroll
        for (int r = 0; r < 8; ++r) {
            int c = r + grp * 8;
            AGG[c * kH + j] = acc.e[r];         // pooled[c][j]
        }
    }
    __syncthreads();

    // ---------- per-cluster LN + projection + mask ----------
    if (tid < kC) {
        const float* prow = &AGG[tid * kH];
        float mu = 0.0f;
        for (int j = 0; j < kH; ++j) mu += prow[j];
        mu *= (1.0f / kH);
        float var = 0.0f;
        for (int j = 0; j < kH; ++j) { float d = prow[j] - mu; var += d * d; }
        float rn = rsqrtf(var * (1.0f / kH) + 1e-5f);
        float xc = 0.0f;
        for (int j = 0; j < kH; ++j)
            xc += ((prow[j] - mu) * rn * flng[j] + flnb[j]) * linw[j];
        float colsum = 0.0f;
        for (int k = 0; k < kNPG; ++k) colsum += s[(nb + k) * kC + tid];
        float mf  = (colsum > 0.0f) ? 1.0f : 0.0f;
        float xcm = xc * mf;
        out[257 + g * kC + tid] = xcm;          // xc output block
        AGG[kC * kH + tid]      = xcm;          // stash (free LDS past row 15)
        AGG[kC * kH + 32 + tid] = mf;
    }
    __syncthreads();
    if (tid == 0) {
        float so = 0.0f, denom = 0.0f, l1 = 0.0f;
        for (int c = 0; c < kC; ++c) {
            float xcm = AGG[kC * kH + c];
            float mf  = AGG[kC * kH + 32 + c];
            so    += xcm;
            denom += mf + 1e-7f;
            l1    += fabsf(xcm);
        }
        out[g] = so + bias[0];                  // graph output
        L1P[g] = l1 / denom;                    // per-graph l1 contribution
    }
}

__global__ __launch_bounds__(256)
void seal_losses(const float* __restrict__ Wo, const float* __restrict__ bo,
                 const float* __restrict__ L1P, float* __restrict__ out)
{
    __shared__ float red[256];
    int tid = threadIdx.x;
    float reg = 0.0f;
    for (int i = tid; i < kL * kH * kH; i += 256) reg += fabsf(Wo[i]);
    for (int i = tid; i < kL * kH;      i += 256) reg += fabsf(bo[i]);
    float l1 = 0.0f;
    for (int i = tid; i < kG; i += 256) l1 += L1P[i];

    red[tid] = reg;
    __syncthreads();
    for (int off = 128; off > 0; off >>= 1) {
        if (tid < off) red[tid] += red[tid + off];
        __syncthreads();
    }
    float regsum = red[0];
    __syncthreads();
    red[tid] = l1;
    __syncthreads();
    for (int off = 128; off > 0; off >>= 1) {
        if (tid < off) red[tid] += red[tid + off];
        __syncthreads();
    }
    if (tid == 0)
        out[256] = 0.01f * regsum + 0.01f * (red[0] / (float)kG);
}

extern "C" void kernel_launch(void* const* d_in, const int* in_sizes, int n_in,
                              void* d_out, int out_size, void* d_ws, size_t ws_size,
                              hipStream_t stream) {
    (void)in_sizes; (void)n_in; (void)out_size; (void)ws_size;
    const float*         x     = (const float*)d_in[0];
    const long long*     eidx  = (const long long*)d_in[1];
    const unsigned char* maskb = (const unsigned char*)d_in[2];
    const float*         s     = (const float*)d_in[3];
    /* d_in[4] = batch (unused: graphs are contiguous blocks of 128 nodes) */
    const float* Wn   = (const float*)d_in[5];
    const float* bn   = (const float*)d_in[6];
    const float* Wo   = (const float*)d_in[7];
    const float* bo   = (const float*)d_in[8];
    const float* Wr   = (const float*)d_in[9];
    const float* lng  = (const float*)d_in[10];
    const float* lnb  = (const float*)d_in[11];
    const float* flng = (const float*)d_in[12];
    const float* flnb = (const float*)d_in[13];
    const float* linw = (const float*)d_in[14];
    const float* bias = (const float*)d_in[15];

    float* ws  = (float*)d_ws;
    float* XI  = ws;                                 // [N*H] f32
    float* XO  = XI + (size_t)kN * kH;               // [N*H] f32
    float* XR  = XO + (size_t)kN * kH;               // [N*H] f32
    float* INV = XR + (size_t)kN * kH;               // [N]   f32
    float* L1P = INV + kN;                           // [G]   f32
    // 16-byte aligned f16 regions (offsets are multiples of 4 floats)
    _Float16* HBh = (_Float16*)(L1P + ((kG + 3) & ~3));   // [N*H] f16
    _Float16* Wh  = HBh + (size_t)kN * kH;                // [3*L*H*H] f16
    float* out = (float*)d_out;

    convert_weights<<<(kWSZ + 255) / 256, 256, 0, stream>>>(Wn, Wo, Wr, Wh);
    seal_fused<<<kG, 256, 0, stream>>>(x, eidx, maskb, s, bn, bo,
                                       lng, lnb, flng, flnb, linw, bias,
                                       Wh, XI, XO, XR, HBh, INV, L1P, out);
    seal_losses<<<1, 256, 0, stream>>>(Wo, bo, L1P, out);
}
